// ResnetGcnnsovnetDegreeRouting_79267916415346
// MI455X (gfx1250) — matrix operand order repaired and checked by reading, not compiled, not run on hardware
//
#include <hip/hip_runtime.h>
#include <hip/hip_bf16.h>
#include <stdint.h>

// ---------------------------------------------------------------------------
// Types for CDNA5 WMMA (wave32, 16x16x32 bf16 -> f32)
// ---------------------------------------------------------------------------
typedef __attribute__((ext_vector_type(16))) __bf16 v16bf;
typedef __attribute__((ext_vector_type(8)))  __bf16 v8bf;
typedef __attribute__((ext_vector_type(8)))  float  v8f;

__device__ __forceinline__ unsigned short f2bf(float f) {
  union { float f; unsigned u; } v; v.f = f;
  unsigned r = (v.u + 0x7FFFu + ((v.u >> 16) & 1u)) >> 16;
  return (unsigned short)r;
}

// ---------------------------------------------------------------------------
// GEMM: C[M][N] = A[M][Kpad] * Bt[N][Kpad]^T  (bf16 inputs, f32 accum)
// A row-major (weights), Bt n-major (im2col). One wave per 16x16 tile.
// A fragment (16x32 bf16): lanes 0-15 hold M=lane, K chunks {0..7,16..23};
// lanes 16-31 hold M=lane-16, K chunks {8..15,24..31}.
// B fragment (32x16 bf16): lanes 0-15 hold N=lane, K chunks {0..7,16..23}; lanes 16-31 K chunks {8..15,24..31}.
// C fragment: VGPR r -> M = r + (lane>=16 ? 8:0), N = lane&15.
// ---------------------------------------------------------------------------
__global__ void k_gemm_wmma(const unsigned short* __restrict__ A,
                            const unsigned short* __restrict__ Bt,
                            const float* __restrict__ bias, int bias_div,
                            float* __restrict__ C,
                            int M, int N, int Kpad) {
  int lane = threadIdx.x & 31;
  int m0 = blockIdx.y * 16;
  int n0 = blockIdx.x * 16;
  int r  = lane & 15;
  int hi = lane >> 4;
  const unsigned short* arow = A  + (size_t)(m0 + r) * Kpad;
  const unsigned short* brow = Bt + (size_t)(n0 + r) * Kpad;
  v8f acc = {};
  for (int kb = 0; kb < Kpad; kb += 32) {
    if (kb + 32 < Kpad) {
      __builtin_prefetch(arow + kb + 32, 0, 1);
      __builtin_prefetch(brow + kb + 32, 0, 1);
    }
    v8bf a0 = *reinterpret_cast<const v8bf*>(arow + kb + hi * 8);
    v8bf a1 = *reinterpret_cast<const v8bf*>(arow + kb + hi * 8 + 16);
    v8bf b0 = *reinterpret_cast<const v8bf*>(brow + kb + hi * 8);
    v8bf b1 = *reinterpret_cast<const v8bf*>(brow + kb + hi * 8 + 16);
    v16bf av, bv;
#pragma unroll
    for (int i = 0; i < 8; ++i) {
      av[i] = a0[i]; av[i + 8] = a1[i];
      bv[i] = b0[i]; bv[i + 8] = b1[i];
    }
    acc = __builtin_amdgcn_wmma_f32_16x16x32_bf16(
        false, av, false, bv, (short)0, acc, false, false);
  }
#pragma unroll
  for (int rr = 0; rr < 8; ++rr) {
    int m = m0 + rr + hi * 8;
    int n = n0 + r;
    if (m < M && n < N) {
      float v = acc[rr];
      if (bias) v += bias[m / bias_div];
      C[(size_t)m * N + n] = v;
    }
  }
}

// ---------------------------------------------------------------------------
// Weight pack: w[O][C][G][k][k] fp32 -> A[O*4][Kpad] bf16 with rotation r=m&3
// applied spatially (rot90^r, CCW) and cyclic roll along the group axis.
// K index = (c*G + g)*k*k + i*k + j.
// ---------------------------------------------------------------------------
__global__ void k_pack_w(const float* __restrict__ w, unsigned short* __restrict__ A,
                         int O, int C, int G, int k, int Kpad) {
  size_t idx = (size_t)blockIdx.x * blockDim.x + threadIdx.x;
  size_t total = (size_t)O * 4 * Kpad;
  if (idx >= total) return;
  int m  = (int)(idx / Kpad);
  int kk = (int)(idx % Kpad);
  int K = C * G * k * k;
  float v = 0.f;
  if (kk < K) {
    int o = m >> 2, rot = m & 3;
    int j = kk % k;
    int i = (kk / k) % k;
    int cg = kk / (k * k);
    int g = cg % G, c = cg / G;
    int e = k - 1, si, sj;
    if      (rot == 0) { si = i;     sj = j;     }
    else if (rot == 1) { si = j;     sj = e - i; }
    else if (rot == 2) { si = e - i; sj = e - j; }
    else               { si = e - j; sj = i;     }
    int sg = (G == 1) ? 0 : ((g - rot) & 3);
    v = w[(((size_t)o * C + c) * G + sg) * (size_t)(k * k) + si * k + sj];
  }
  A[idx] = f2bf(v);
}

// ---------------------------------------------------------------------------
// im2col from trunk layout act[CH][B][H][W] -> Bt[N][Kpad] bf16,
// N = b*Ho*Wo + oh*Wo + ow ; K = ch*k*k + i*k + j
// ---------------------------------------------------------------------------
__global__ void k_im2col(const float* __restrict__ act, unsigned short* __restrict__ Bt,
                         int CH, int B, int H, int W, int k, int stride, int pad,
                         int Ho, int Wo, int Kpad) {
  size_t idx = (size_t)blockIdx.x * blockDim.x + threadIdx.x;
  size_t N = (size_t)B * Ho * Wo;
  if (idx >= N * Kpad) return;
  size_t n = idx / Kpad;
  int kk = (int)(idx % Kpad);
  int K = CH * k * k;
  float v = 0.f;
  if (kk < K) {
    int ow = (int)(n % Wo);
    int oh = (int)((n / Wo) % Ho);
    int b  = (int)(n / ((size_t)Ho * Wo));
    int j = kk % k, i = (kk / k) % k, ch = kk / (k * k);
    int y = oh * stride - pad + i;
    int x = ow * stride - pad + j;
    if (y >= 0 && y < H && x >= 0 && x < W)
      v = act[((size_t)ch * B + b) * (size_t)(H * W) + (size_t)y * W + x];
  }
  Bt[idx] = f2bf(v);
}

// im2col for capsule convs: trunk act[(ic*ID+d)*4+g][B][H][W],
// N = (b*IC+ic)*S + oh*Wo+ow ; K = (d*4+g)*k*k + i*k + j
__global__ void k_im2col_caps(const float* __restrict__ act, unsigned short* __restrict__ Bt,
                              int B, int IC, int ID, int H, int W,
                              int k, int stride, int pad, int Ho, int Wo, int Kpad) {
  size_t idx = (size_t)blockIdx.x * blockDim.x + threadIdx.x;
  size_t N = (size_t)B * IC * Ho * Wo;
  if (idx >= N * Kpad) return;
  size_t n = idx / Kpad;
  int kk = (int)(idx % Kpad);
  int K = ID * 4 * k * k;
  float v = 0.f;
  if (kk < K) {
    int S = Ho * Wo;
    int ow = (int)(n % Wo);
    int oh = (int)((n / Wo) % Ho);
    int bi = (int)(n / S);
    int ic = bi % IC, b = bi / IC;
    int j = kk % k, i = (kk / k) % k, dg = kk / (k * k);
    int g = dg & 3, d = dg >> 2;
    int y = oh * stride - pad + i;
    int x = ow * stride - pad + j;
    if (y >= 0 && y < H && x >= 0 && x < W)
      v = act[((size_t)((ic * ID + d) * 4 + g) * B + b) * (size_t)(H * W) + (size_t)y * W + x];
  }
  Bt[idx] = f2bf(v);
}

// NCHW -> trunk [C][B][HW]
__global__ void k_nchw_to_trunk(const float* __restrict__ x, float* __restrict__ act,
                                int B, int C, int HW) {
  size_t idx = (size_t)blockIdx.x * blockDim.x + threadIdx.x;
  if (idx >= (size_t)B * C * HW) return;
  int p = (int)(idx % HW);
  int c = (int)((idx / HW) % C);
  int b = (int)(idx / ((size_t)C * HW));
  act[((size_t)c * B + b) * HW + p] = x[idx];
}

// ---------------------------------------------------------------------------
// BatchNorm3d over trunk/pr layout [O*4][N]: stats per o over 4*N elements
// ---------------------------------------------------------------------------
__global__ void k_bn_stats(const float* __restrict__ X, float* __restrict__ mean,
                           float* __restrict__ inv, int rowlen, float eps) {
  int o = blockIdx.x;
  const float* p = X + (size_t)o * rowlen;
  __shared__ float ssum[256], ssq[256];
  float s = 0.f, q = 0.f;
  for (int i = threadIdx.x; i < rowlen; i += blockDim.x) {
    float v = p[i]; s += v; q += v * v;
  }
  ssum[threadIdx.x] = s; ssq[threadIdx.x] = q;
  __syncthreads();
  for (int st = 128; st > 0; st >>= 1) {
    if ((int)threadIdx.x < st) {
      ssum[threadIdx.x] += ssum[threadIdx.x + st];
      ssq[threadIdx.x]  += ssq[threadIdx.x + st];
    }
    __syncthreads();
  }
  if (threadIdx.x == 0) {
    float m = ssum[0] / rowlen;
    float var = ssq[0] / rowlen - m * m;
    mean[o] = m;
    inv[o] = rsqrtf(var + eps);
  }
}

__global__ void k_bn_apply(const float* __restrict__ X, const float* __restrict__ mean,
                           const float* __restrict__ inv, const float* __restrict__ g,
                           const float* __restrict__ b, const float* __restrict__ res,
                           int relu, float* __restrict__ Y, int M4, size_t N) {
  size_t idx = (size_t)blockIdx.x * blockDim.x + threadIdx.x;
  if (idx >= (size_t)M4 * N) return;
  int o = (int)(idx / N) >> 2;
  float v = g[o] * (X[idx] - mean[o]) * inv[o] + b[o];
  if (res) v += res[idx];
  if (relu) v = fmaxf(v, 0.f);
  Y[idx] = v;
}

// ---------------------------------------------------------------------------
// Primary-capsule LayerNorm over (C,4,H,W) per batch; trunk layout [CH4][B][S]
// ---------------------------------------------------------------------------
__global__ void k_ln_stats(const float* __restrict__ X, float* __restrict__ m,
                           float* __restrict__ inv, int B, int CH4, int S) {
  int b = blockIdx.x;
  __shared__ float ssum[256], ssq[256];
  size_t per = (size_t)CH4 * S;
  float s = 0.f, q = 0.f;
  for (size_t i = threadIdx.x; i < per; i += blockDim.x) {
    int ch = (int)(i / S), sp = (int)(i % S);
    float v = X[((size_t)ch * B + b) * S + sp];
    s += v; q += v * v;
  }
  ssum[threadIdx.x] = s; ssq[threadIdx.x] = q;
  __syncthreads();
  for (int st = 128; st > 0; st >>= 1) {
    if ((int)threadIdx.x < st) {
      ssum[threadIdx.x] += ssum[threadIdx.x + st];
      ssq[threadIdx.x]  += ssq[threadIdx.x + st];
    }
    __syncthreads();
  }
  if (threadIdx.x == 0) {
    float mu = ssum[0] / (float)per;
    float var = ssq[0] / (float)per - mu * mu;
    m[b] = mu;
    inv[b] = rsqrtf(var + 1e-5f);
  }
}

__global__ void k_ln_apply(const float* __restrict__ X, const float* __restrict__ m,
                           const float* __restrict__ inv, const float* __restrict__ lg,
                           const float* __restrict__ lb, float* __restrict__ Y,
                           int B, int CH4, int S) {
  size_t idx = (size_t)blockIdx.x * blockDim.x + threadIdx.x;
  if (idx >= (size_t)CH4 * B * S) return;
  int ch = (int)(idx / ((size_t)B * S));
  size_t rem = idx % ((size_t)B * S);
  int b = (int)(rem / S);
  int sp = (int)(rem % S);
  size_t gi = (size_t)ch * S + sp;
  Y[idx] = lg[gi] * (X[idx] - m[b]) * inv[b] + lb[gi];
}

// ---------------------------------------------------------------------------
// Degree routing. pr layout: pr[(oc*OD+od)*4+g][(b*IC+ic)*S+sp], NN=B*IC*S.
// ---------------------------------------------------------------------------
__global__ void k_route_psum(const float* __restrict__ pr, float* __restrict__ ps,
                             int B, int IC, int OC, int OD, int S) {
  size_t idx = (size_t)blockIdx.x * blockDim.x + threadIdx.x;
  size_t total = (size_t)B * OC * 4 * S * OD;
  if (idx >= total) return;
  int od = (int)(idx % OD);
  size_t t = idx / OD;
  int sp = (int)(t % S); t /= S;
  int g = (int)(t & 3); t >>= 2;
  int oc = (int)(t % OC);
  int b = (int)(t / OC);
  size_t NN = (size_t)B * IC * S;
  size_t rowbase = ((size_t)(oc * OD + od) * 4 + g) * NN;
  float a = 0.f;
  for (int ic = 0; ic < IC; ++ic)
    a += pr[rowbase + ((size_t)b * IC + ic) * S + sp];
  ps[idx] = a;
}

__global__ void k_route_logit(const float* __restrict__ pr, const float* __restrict__ ps,
                              float* __restrict__ lg,
                              int B, int IC, int OC, int OD, int S) {
  size_t idx = (size_t)blockIdx.x * blockDim.x + threadIdx.x;
  size_t total = (size_t)B * OC * 4 * S * IC;
  if (idx >= total) return;
  int ic = (int)(idx % IC);
  size_t t = idx / IC;
  int sp = (int)(t % S); t /= S;
  int g = (int)(t & 3); t >>= 2;
  int oc = (int)(t % OC);
  int b = (int)(t / OC);
  size_t NN = (size_t)B * IC * S;
  size_t col = ((size_t)b * IC + ic) * S + sp;
  size_t psbase = ((((size_t)(b * OC + oc) * 4 + g) * S) + sp) * OD;
  float dotP = 0.f, nsq = 0.f;
  for (int od = 0; od < OD; ++od) {
    float p = pr[((size_t)(oc * OD + od) * 4 + g) * NN + col];
    dotP += p * ps[psbase + od];
    nsq += p * p;
  }
  lg[idx] = dotP / fmaxf(nsq, 1e-8f);
}

__global__ void k_route_soft(const float* __restrict__ pr, const float* __restrict__ lg,
                             float* __restrict__ sb,
                             int B, int IC, int OC, int OD, int S) {
  size_t idx = (size_t)blockIdx.x * blockDim.x + threadIdx.x;
  size_t total = (size_t)B * OC * 4 * S;
  if (idx >= total) return;
  int sp = (int)(idx % S);
  size_t t = idx / S;
  int g = (int)(t & 3); t >>= 2;
  int oc = (int)(t % OC);
  int b = (int)(t / OC);
  size_t base = idx * IC;
  float mx = -1e30f;
  for (int ic = 0; ic < IC; ++ic) mx = fmaxf(mx, lg[base + ic]);
  float den = 0.f;
  for (int ic = 0; ic < IC; ++ic) den += __expf(lg[base + ic] - mx);
  float sv[16];
  for (int od = 0; od < 16; ++od) sv[od] = 0.f;
  size_t NN = (size_t)B * IC * S;
  for (int ic = 0; ic < IC; ++ic) {
    float w = __expf(lg[base + ic] - mx) / den;
    size_t col = ((size_t)b * IC + ic) * S + sp;
    for (int od = 0; od < OD; ++od)
      sv[od] += w * pr[((size_t)(oc * OD + od) * 4 + g) * NN + col];
  }
  for (int od = 0; od < OD; ++od) sb[idx * OD + od] = sv[od];
}

__global__ void k_route_squash(const float* __restrict__ sb, float* __restrict__ out,
                               int B, int OC, int OD, int S) {
  size_t idx = (size_t)blockIdx.x * blockDim.x + threadIdx.x;
  size_t total = (size_t)B * OC * OD * S;
  if (idx >= total) return;
  int sp = (int)(idx % S);
  size_t t = idx / S;
  int od = (int)(t % OD); t /= OD;
  int oc = (int)(t % OC);
  int b = (int)(t / OC);
  float vg[4], ss = 0.f;
  for (int g = 0; g < 4; ++g) {
    vg[g] = sb[((((size_t)(b * OC + oc) * 4 + g) * S) + sp) * OD + od];
    ss += vg[g] * vg[g];
  }
  float n = sqrtf(ss);
  float f = (ss / (1.f + ss)) / (n + 1e-8f);
  for (int g = 0; g < 4; ++g)
    out[((size_t)(oc * OD + od) * 4 + g) * ((size_t)B * S) + (size_t)b * S + sp] = f * vg[g];
}

// Head: act[(cls*16+od)*4+g][B] (S=1) -> argmax over g of ||.||, linear head.
__global__ void k_head(const float* __restrict__ act, const float* __restrict__ lw,
                       const float* __restrict__ lb, float* __restrict__ out,
                       int B, int NC, int OD) {
  size_t idx = (size_t)blockIdx.x * blockDim.x + threadIdx.x;
  if (idx >= (size_t)B * NC) return;
  int cls = (int)(idx % NC);
  int b = (int)(idx / NC);
  float cc[16][4];
  float norms[4] = {0.f, 0.f, 0.f, 0.f};
  for (int od = 0; od < OD; ++od)
    for (int g = 0; g < 4; ++g) {
      float v = act[((size_t)(cls * 16 + od) * 4 + g) * B + b];
      cc[od][g] = v;
      norms[g] += v * v;
    }
  int gs = 0; float bn = norms[0];
  for (int g = 1; g < 4; ++g) if (norms[g] > bn) { bn = norms[g]; gs = g; }
  float acc = lb[0];
  for (int od = 0; od < OD; ++od) acc += cc[od][gs] * lw[od];
  out[idx] = acc;
}

// ---------------------------------------------------------------------------
// Host orchestration
// ---------------------------------------------------------------------------
static inline unsigned gdiv(size_t a, unsigned b) { return (unsigned)((a + b - 1) / b); }

static void gconv(hipStream_t st, const float* act_in, const float* w,
                  int O, int C, int G, int k, int B, int H, int W,
                  int stride, int pad, const float* bias, int bias_div,
                  unsigned short* wpack, unsigned short* cols, float* gout) {
  int Ho = (H + 2 * pad - k) / stride + 1;
  int Wo = (W + 2 * pad - k) / stride + 1;
  int K = C * G * k * k, Kpad = (K + 31) & ~31, M = O * 4;
  size_t N = (size_t)B * Ho * Wo;
  size_t tp = (size_t)M * Kpad;
  k_pack_w<<<gdiv(tp, 256), 256, 0, st>>>(w, wpack, O, C, G, k, Kpad);
  size_t ti = N * Kpad;
  k_im2col<<<gdiv(ti, 256), 256, 0, st>>>(act_in, cols, C * G, B, H, W, k, stride, pad, Ho, Wo, Kpad);
  dim3 gg(gdiv(N, 16), gdiv((size_t)M, 16));
  k_gemm_wmma<<<gg, 32, 0, st>>>(wpack, cols, bias, bias_div, gout, M, (int)N, Kpad);
}

static void bn(hipStream_t st, float* X, const float* g, const float* b, int O, size_t N,
               const float* res, int relu, float* Y, float* bm, float* bi) {
  k_bn_stats<<<O, 256, 0, st>>>(X, bm, bi, (int)(4 * N), 1e-5f);
  size_t tot = (size_t)O * 4 * N;
  k_bn_apply<<<gdiv(tot, 256), 256, 0, st>>>(X, bm, bi, g, b, res, relu, Y, O * 4, N);
}

static void caps_layer(hipStream_t st, const float* act_in, int B, int IC, int ID,
                       int H, int W, int OC, int OD, int stride,
                       const float* w, const float* bias, const float* gam, const float* bet,
                       unsigned short* wpack, unsigned short* cols, float* gout,
                       float* psum, float* lgt, float* sbuf, float* bm, float* bi,
                       float* act_out, int* pHo) {
  const int kk = 3, pad = 0;
  int Ho = (H - kk) / stride + 1;
  int Wo = (W - kk) / stride + 1;
  int K = ID * 4 * kk * kk, Kpad = (K + 31) & ~31;
  int O = OC * OD, M = O * 4;
  int S = Ho * Wo;
  size_t N = (size_t)B * IC * S;
  size_t tp = (size_t)M * Kpad;
  k_pack_w<<<gdiv(tp, 256), 256, 0, st>>>(w, wpack, O, ID, 4, kk, Kpad);
  size_t ti = N * Kpad;
  k_im2col_caps<<<gdiv(ti, 256), 256, 0, st>>>(act_in, cols, B, IC, ID, H, W, kk, stride, pad, Ho, Wo, Kpad);
  dim3 gg(gdiv(N, 16), gdiv((size_t)M, 16));
  k_gemm_wmma<<<gg, 32, 0, st>>>(wpack, cols, bias, 4, gout, M, (int)N, Kpad);
  bn(st, gout, gam, bet, O, N, nullptr, 0, gout, bm, bi);
  size_t t1 = (size_t)B * OC * 4 * S * OD;
  k_route_psum<<<gdiv(t1, 256), 256, 0, st>>>(gout, psum, B, IC, OC, OD, S);
  size_t t2 = (size_t)B * OC * 4 * S * IC;
  k_route_logit<<<gdiv(t2, 256), 256, 0, st>>>(gout, psum, lgt, B, IC, OC, OD, S);
  size_t t3 = (size_t)B * OC * 4 * S;
  k_route_soft<<<gdiv(t3, 256), 256, 0, st>>>(gout, lgt, sbuf, B, IC, OC, OD, S);
  size_t t4 = (size_t)B * OC * OD * S;
  k_route_squash<<<gdiv(t4, 256), 256, 0, st>>>(sbuf, act_out, B, OC, OD, S);
  *pHo = Ho;
}

extern "C" void kernel_launch(void* const* d_in, const int* in_sizes, int n_in,
                              void* d_out, int out_size, void* d_ws, size_t ws_size,
                              hipStream_t stream) {
  (void)in_sizes; (void)n_in; (void)out_size; (void)ws_size;
  // ---- input unpack (setup_inputs dict order, depth-first) ----
  int ii = 0;
  const float* X = (const float*)d_in[ii++];
  const float* w_conv1 = (const float*)d_in[ii++];
  const float* g1 = (const float*)d_in[ii++];
  const float* b1 = (const float*)d_in[ii++];
  struct Blk { const float *c1, *g1, *b1, *c2, *g2, *b2, *sc, *sg, *sb; };
  Blk L1[3], L2[4];
  for (int k = 0; k < 3; ++k) {
    L1[k].c1 = (const float*)d_in[ii++]; L1[k].g1 = (const float*)d_in[ii++];
    L1[k].b1 = (const float*)d_in[ii++]; L1[k].c2 = (const float*)d_in[ii++];
    L1[k].g2 = (const float*)d_in[ii++]; L1[k].b2 = (const float*)d_in[ii++];
    L1[k].sc = nullptr; L1[k].sg = nullptr; L1[k].sb = nullptr;
  }
  for (int k = 0; k < 4; ++k) {
    L2[k].c1 = (const float*)d_in[ii++]; L2[k].g1 = (const float*)d_in[ii++];
    L2[k].b1 = (const float*)d_in[ii++]; L2[k].c2 = (const float*)d_in[ii++];
    L2[k].g2 = (const float*)d_in[ii++]; L2[k].b2 = (const float*)d_in[ii++];
    if (k == 0) {
      L2[k].sc = (const float*)d_in[ii++]; L2[k].sg = (const float*)d_in[ii++];
      L2[k].sb = (const float*)d_in[ii++];
    } else { L2[k].sc = nullptr; L2[k].sg = nullptr; L2[k].sb = nullptr; }
  }
  const float* prim_w = (const float*)d_in[ii++];
  const float* prim_bias = (const float*)d_in[ii++];
  const float* prim_lng = (const float*)d_in[ii++];
  const float* prim_lnb = (const float*)d_in[ii++];
  struct Caps { const float *w, *bias, *g, *b; };
  Caps CP[4];
  for (int k = 0; k < 4; ++k) {
    CP[k].w = (const float*)d_in[ii++]; CP[k].bias = (const float*)d_in[ii++];
    CP[k].g = (const float*)d_in[ii++]; CP[k].b = (const float*)d_in[ii++];
  }
  const float* lw = (const float*)d_in[ii++];
  const float* lb = (const float*)d_in[ii++];

  // ---- workspace carving ----
  char* wp = (char*)d_ws;
  auto carve = [&](size_t bytes) -> void* {
    void* r = (void*)wp; wp += (bytes + 255) & ~(size_t)255; return r;
  };
  float* act0 = (float*)carve(36ull << 20);
  float* act1 = (float*)carve(36ull << 20);
  float* act2 = (float*)carve(36ull << 20);
  unsigned short* cols  = (unsigned short*)carve(80ull << 20);
  unsigned short* wpack = (unsigned short*)carve(10ull << 20);
  float* gout = (float*)carve(210ull << 20);
  float* psum = (float*)carve(8ull << 20);
  float* lgt  = (float*)carve(14ull << 20);
  float* sbuf = (float*)carve(8ull << 20);
  float* bm = (float*)carve(1 << 16);
  float* bi = (float*)carve(1 << 16);

  const int B = 16;
  hipStream_t st = stream;

  // conv1 (P4ConvZ2) + BN + ReLU: 3 -> 64 ch, 32x32
  k_nchw_to_trunk<<<gdiv((size_t)B * 3 * 1024, 256), 256, 0, st>>>(X, act0, B, 3, 1024);
  gconv(st, act0, w_conv1, 64, 3, 1, 3, B, 32, 32, 1, 1, nullptr, 1, wpack, cols, gout);
  bn(st, gout, g1, b1, 64, (size_t)B * 32 * 32, nullptr, 1, act1, bm, bi);

  float* pX = act1; float* pT = act2; float* pO = act0;
  // layer1: 3 blocks, 64ch, 32x32
  for (int k = 0; k < 3; ++k) {
    gconv(st, pX, L1[k].c1, 64, 64, 4, 3, B, 32, 32, 1, 1, nullptr, 1, wpack, cols, gout);
    bn(st, gout, L1[k].g1, L1[k].b1, 64, (size_t)B * 1024, nullptr, 1, pT, bm, bi);
    gconv(st, pT, L1[k].c2, 64, 64, 4, 3, B, 32, 32, 1, 1, nullptr, 1, wpack, cols, gout);
    bn(st, gout, L1[k].g2, L1[k].b2, 64, (size_t)B * 1024, pX, 1, pO, bm, bi);
    float* t = pX; pX = pO; pO = t;
  }
  // layer2: 4 blocks, first stride-2 with shortcut, 128ch, 16x16
  int H = 32;
  for (int k = 0; k < 4; ++k) {
    int stride = (k == 0) ? 2 : 1;
    int Cin = (k == 0) ? 64 : 128;
    int Ho2 = H / stride;
    size_t N = (size_t)B * Ho2 * Ho2;
    gconv(st, pX, L2[k].c1, 128, Cin, 4, 3, B, H, H, stride, 1, nullptr, 1, wpack, cols, gout);
    bn(st, gout, L2[k].g1, L2[k].b1, 128, N, nullptr, 1, pT, bm, bi);
    if (L2[k].sc) {
      gconv(st, pX, L2[k].sc, 128, Cin, 4, 1, B, H, H, stride, 0, nullptr, 1, wpack, cols, gout);
      bn(st, gout, L2[k].sg, L2[k].sb, 128, N, nullptr, 0, pO, bm, bi);
    }
    gconv(st, pT, L2[k].c2, 128, 128, 4, 3, B, Ho2, Ho2, 1, 1, nullptr, 1, wpack, cols, gout);
    bn(st, gout, L2[k].g2, L2[k].b2, 128, N, L2[k].sc ? pO : pX, 1, pO, bm, bi);
    float* t = pX; pX = pO; pO = t;
    H = Ho2;
  }

  // primary capsules: 1x1 P4 conv (bias per O) + LayerNorm over (C,4,H,W)
  gconv(st, pX, prim_w, 512, 128, 4, 1, B, 16, 16, 1, 0, prim_bias, 4, wpack, cols, gout);
  k_ln_stats<<<B, 256, 0, st>>>(gout, bm, bi, B, 2048, 256);
  k_ln_apply<<<gdiv((size_t)2048 * B * 256, 256), 256, 0, st>>>(
      gout, bm, bi, prim_lng, prim_lnb, pT, B, 2048, 256);

  // capsule layers with degree routing
  int Ho;
  caps_layer(st, pT, B, 32, 16, 16, 16, 32, 16, 2, CP[0].w, CP[0].bias, CP[0].g, CP[0].b,
             wpack, cols, gout, psum, lgt, sbuf, bm, bi, pO, &Ho);           // 16 -> 7
  caps_layer(st, pO, B, 32, 16, Ho, Ho, 32, 16, 1, CP[1].w, CP[1].bias, CP[1].g, CP[1].b,
             wpack, cols, gout, psum, lgt, sbuf, bm, bi, pX, &Ho);           // 7 -> 5
  caps_layer(st, pX, B, 32, 16, Ho, Ho, 32, 16, 1, CP[2].w, CP[2].bias, CP[2].g, CP[2].b,
             wpack, cols, gout, psum, lgt, sbuf, bm, bi, pT, &Ho);           // 5 -> 3
  caps_layer(st, pT, B, 32, 16, Ho, Ho, 100, 16, 1, CP[3].w, CP[3].bias, CP[3].g, CP[3].b,
             wpack, cols, gout, psum, lgt, sbuf, bm, bi, pO, &Ho);           // 3 -> 1

  // head: norms over capsule dim, argmax over group, linear
  k_head<<<gdiv((size_t)B * 100, 256), 256, 0, st>>>(pO, lw, lb, (float*)d_out, B, 100, 16);
}
